// BistrideGraphMessagePassing_25924422598772
// MI455X (gfx1250) — matrix operand order repaired and checked by reading, not compiled, not run on hardware
//
#include <hip/hip_runtime.h>
#include <math.h>

// ---------------------------------------------------------------------------
// BistrideGraphMessagePassing on MI455X (gfx1250, wave32, WMMA bf16)
//
// Fused WMMA MLP kernels with fragment-major weight staging in LDS:
//   - B fragments: one contiguous 32B/lane read (2x ds_load_b128), no packing
//   - A fragments: contiguous row-major reads (2x ds_load_b128)
//   - 4 row-tiles per wave per B fragment (4 independent WMMA chains)
// ---------------------------------------------------------------------------

typedef __bf16 bf16;
typedef __attribute__((ext_vector_type(16))) __bf16 v16bf;
typedef __attribute__((ext_vector_type(8)))  float  v8f;

#define LAT 128          // latent width
#define MT  64           // rows (edges/nodes) per block tile
#define TPB 256          // 8 wave32s per block
#define RG  (TPB / 8)    // 32 row-groups of 8 threads for build/LN/epilogue
constexpr int K0P_EDGE = 288;   // 260 (=2*128+3+1) padded to multiple of 32
constexpr int K0P_NODE = 256;   // 2*128

// shared-memory footprint (must match the carve inside the kernels)
constexpr size_t smem_bytes(int k0p) {
    return (size_t)(k0p * LAT + 2 * LAT * LAT + MT * k0p + 2 * MT * LAT) * 2  // bf16: W0f,W1f,W2f,X,H1,H2
         + (size_t)(5 * LAT + MT * LAT) * 4                                   // f32: b0,b1,b2,g,be,H3
         + (size_t)(2 * MT) * 4;                                              // int: jrow,vrow
}

__device__ __forceinline__ v8f wmma_bf16(v16bf a, v16bf b, v8f c) {
    return __builtin_amdgcn_wmma_f32_16x16x32_bf16(false, a, false, b, (short)0, c, false, false);
}

// A fragment (16x32 bf16, M x K) from row-major LDS [rows][ldx].
// ISA layout: lanes 0-15 -> M=0..15 (K groups 0..7,16..23), lanes 16-31 -> +8 K offset.
// Two contiguous 16B chunks per lane -> 2x ds_load_b128.
__device__ __forceinline__ v16bf load_a(const bf16* X, int ldx, int mbase, int kbase, int lane) {
    const int half = lane >> 4, m = lane & 15;
    const bf16* row = X + (size_t)(mbase + m) * ldx + kbase + half * 8;
    v16bf a;
#pragma unroll
    for (int e = 0; e < 16; ++e) a[e] = (e < 8) ? row[e] : row[8 + e];
    return a;
}

// B fragment from fragment-major LDS weights: Wf[kb5][cb][lane][e] (e contiguous).
// One 32B contiguous read per lane -> 2x ds_load_b128, zero packing VALU.
__device__ __forceinline__ v16bf load_bf(const bf16* Wf, int kb5, int cb, int lane) {
    const bf16* p = Wf + (((size_t)kb5 * 8 + cb) * 32 + lane) * 16;
    v16bf b;
#pragma unroll
    for (int e = 0; e < 16; ++e) b[e] = p[e];
    return b;
}

// Stage a row-major [K][LAT] f32 weight into fragment-major bf16 LDS layout.
// B-fragment mapping: lane = ((klocal>>3)&1)*16 + (n&15), element e = (klocal&7) + (klocal>>4)*8, cb = n>>4.
__device__ __forceinline__ void stage_weight(bf16* Wf, const float* w, int K, int Kp) {
    for (int idx = threadIdx.x; idx < Kp * LAT; idx += TPB) {
        const int k = idx / LAT, n = idx % LAT;
        const float v = (k < K) ? w[(size_t)k * LAT + n] : 0.0f;
        const int kb5 = k >> 5, kl = k & 31;
        const int lane = (((kl >> 3) & 1) << 4) | (n & 15);
        const int e = (kl & 7) | ((kl >> 4) << 3);
        const int cb = n >> 4;
        Wf[(((size_t)kb5 * 8 + cb) * 32 + lane) * 16 + e] = (bf16)v;
    }
}

// C/D layout: VGPR r -> row (r + 8*(lane>>4)), col (lane&15).
__device__ __forceinline__ void store_act_bf16(bf16* H, int mbase, int cb, int lane,
                                               const float* bias, v8f acc) {
    const int half = lane >> 4, n = cb * 16 + (lane & 15);
#pragma unroll
    for (int r = 0; r < 8; ++r) {
        int m = mbase + r + 8 * half;
        float v = acc[r] + bias[n];
        v = v / (1.0f + __expf(-v));          // SiLU
        H[m * LAT + n] = (bf16)v;
    }
}

__device__ __forceinline__ void store_act_f32(float* H, int mbase, int cb, int lane,
                                              const float* bias, v8f acc) {
    const int half = lane >> 4, n = cb * 16 + (lane & 15);
#pragma unroll
    for (int r = 0; r < 8; ++r) {
        int m = mbase + r + 8 * half;
        H[m * LAT + n] = acc[r] + bias[n];
    }
}

// 3-layer MLP on a 64-row tile: X[MT][Kp] @ W0 -> SiLU -> @W1 -> SiLU -> @W2 (+biases)
// Each of the 8 waves owns one 16-col block and 4 row-tiles (4 WMMA chains).
__device__ __forceinline__ void mlp3(const bf16* X, int Kp,
                                     const bf16* W0f, const bf16* W1f, const bf16* W2f,
                                     const float* b0, const float* b1, const float* b2,
                                     bf16* H1, bf16* H2, float* H3) {
    const int tid = threadIdx.x, lane = tid & 31, cb = tid >> 5;
    {
        v8f c0 = {}, c1 = {}, c2 = {}, c3 = {};
        for (int kb = 0; kb < Kp; kb += 32) {
            v16bf b = load_bf(W0f, kb >> 5, cb, lane);
            c0 = wmma_bf16(load_a(X, Kp, 0,  kb, lane), b, c0);
            c1 = wmma_bf16(load_a(X, Kp, 16, kb, lane), b, c1);
            c2 = wmma_bf16(load_a(X, Kp, 32, kb, lane), b, c2);
            c3 = wmma_bf16(load_a(X, Kp, 48, kb, lane), b, c3);
        }
        store_act_bf16(H1, 0,  cb, lane, b0, c0);
        store_act_bf16(H1, 16, cb, lane, b0, c1);
        store_act_bf16(H1, 32, cb, lane, b0, c2);
        store_act_bf16(H1, 48, cb, lane, b0, c3);
    }
    __syncthreads();
    {
        v8f c0 = {}, c1 = {}, c2 = {}, c3 = {};
#pragma unroll
        for (int kb = 0; kb < LAT; kb += 32) {
            v16bf b = load_bf(W1f, kb >> 5, cb, lane);
            c0 = wmma_bf16(load_a(H1, LAT, 0,  kb, lane), b, c0);
            c1 = wmma_bf16(load_a(H1, LAT, 16, kb, lane), b, c1);
            c2 = wmma_bf16(load_a(H1, LAT, 32, kb, lane), b, c2);
            c3 = wmma_bf16(load_a(H1, LAT, 48, kb, lane), b, c3);
        }
        store_act_bf16(H2, 0,  cb, lane, b1, c0);
        store_act_bf16(H2, 16, cb, lane, b1, c1);
        store_act_bf16(H2, 32, cb, lane, b1, c2);
        store_act_bf16(H2, 48, cb, lane, b1, c3);
    }
    __syncthreads();
    {
        v8f c0 = {}, c1 = {}, c2 = {}, c3 = {};
#pragma unroll
        for (int kb = 0; kb < LAT; kb += 32) {
            v16bf b = load_bf(W2f, kb >> 5, cb, lane);
            c0 = wmma_bf16(load_a(H2, LAT, 0,  kb, lane), b, c0);
            c1 = wmma_bf16(load_a(H2, LAT, 16, kb, lane), b, c1);
            c2 = wmma_bf16(load_a(H2, LAT, 32, kb, lane), b, c2);
            c3 = wmma_bf16(load_a(H2, LAT, 48, kb, lane), b, c3);
        }
        store_act_f32(H3, 0,  cb, lane, b2, c0);
        store_act_f32(H3, 16, cb, lane, b2, c1);
        store_act_f32(H3, 32, cb, lane, b2, c2);
        store_act_f32(H3, 48, cb, lane, b2, c3);
    }
}

// LayerNorm over 128 cols; 8 consecutive threads per row, shfl reduction (wave32).
__device__ __forceinline__ void layernorm_rows(float* H3, const float* g, const float* be) {
    const int tid = threadIdx.x;
    const int sub = tid & 7;
    for (int row = tid >> 3; row < MT; row += RG) {
        float* r = H3 + (size_t)row * LAT;
        float s = 0.f, ss = 0.f;
#pragma unroll
        for (int c = sub * 16; c < sub * 16 + 16; ++c) { float v = r[c]; s += v; ss += v * v; }
#pragma unroll
        for (int m = 1; m < 8; m <<= 1) { s += __shfl_xor(s, m, 8); ss += __shfl_xor(ss, m, 8); }
        const float mu = s * (1.0f / LAT);
        const float var = ss * (1.0f / LAT) - mu * mu;
        const float inv = rsqrtf(var + 1e-5f);
#pragma unroll
        for (int c = sub * 16; c < sub * 16 + 16; ++c)
            r[c] = (r[c] - mu) * inv * g[c] + be[c];
    }
}

// ------------------------- fused edge MLP + segment_sum ---------------------
__global__ __launch_bounds__(TPB)
void edge_mlp_kernel(const float* __restrict__ x, const float* __restrict__ pos,
                     const int* __restrict__ gi, const int* __restrict__ gj,
                     const float* __restrict__ w0, const float* __restrict__ b0,
                     const float* __restrict__ w1, const float* __restrict__ b1,
                     const float* __restrict__ w2, const float* __restrict__ b2,
                     const float* __restrict__ gamma, const float* __restrict__ beta,
                     float* __restrict__ aggr, int E) {
    const int K0 = 2 * LAT + 4;          // 260 real input features
    const int Kp = K0P_EDGE;             // padded
    extern __shared__ char smem[];
    bf16* W0f = (bf16*)smem;
    bf16* W1f = W0f + (size_t)Kp * LAT;
    bf16* W2f = W1f + (size_t)LAT * LAT;
    bf16* X   = W2f + (size_t)LAT * LAT;
    bf16* H1  = X + (size_t)MT * Kp;
    bf16* H2  = H1 + (size_t)MT * LAT;
    float* b0s = (float*)(H2 + (size_t)MT * LAT);
    float* b1s = b0s + LAT;
    float* b2s = b1s + LAT;
    float* gs  = b2s + LAT;
    float* bes = gs + LAT;
    float* H3  = bes + LAT;
    int* jrow = (int*)(H3 + (size_t)MT * LAT);
    int* vrow = jrow + MT;

    const int tid = threadIdx.x;
    stage_weight(W0f, w0, K0, Kp);
    stage_weight(W1f, w1, LAT, LAT);
    stage_weight(W2f, w2, LAT, LAT);
    for (int c = tid; c < LAT; c += TPB) {
        b0s[c] = b0[c]; b1s[c] = b1[c]; b2s[c] = b2[c];
        gs[c] = gamma[c]; bes[c] = beta[c];
    }
    __syncthreads();

    const int ntiles = (E + MT - 1) / MT;
    const int sub = tid & 7;
    for (int t = blockIdx.x; t < ntiles; t += gridDim.x) {
        for (int row = tid >> 3; row < MT; row += RG) {
            const int e = t * MT + row;
            const bool valid = (e < E);
            int i = 0, j = 0;
            if (valid) { i = gi[e]; j = gj[e]; }
            if (sub == 0) { jrow[row] = j; vrow[row] = valid ? 1 : 0; }
            float d0 = 0.f, d1 = 0.f, d2 = 0.f, nrm = 0.f;
            if (valid) {
                d0 = pos[i * 3 + 0] - pos[j * 3 + 0];
                d1 = pos[i * 3 + 1] - pos[j * 3 + 1];
                d2 = pos[i * 3 + 2] - pos[j * 3 + 2];
                nrm = sqrtf(d0 * d0 + d1 * d1 + d2 * d2);
            }
            bf16* xr = X + (size_t)row * Kp;
            for (int c = sub; c < Kp; c += 8) {
                float v = 0.f;
                if (valid) {
                    if (c == 0) v = d0;
                    else if (c == 1) v = d1;
                    else if (c == 2) v = d2;
                    else if (c == 3) v = nrm;
                    else if (c < 4 + LAT) v = x[(size_t)i * LAT + (c - 4)];
                    else if (c < 4 + 2 * LAT) v = x[(size_t)j * LAT + (c - 4 - LAT)];
                }
                xr[c] = (bf16)v;
            }
        }
        __syncthreads();
        mlp3(X, Kp, W0f, W1f, W2f, b0s, b1s, b2s, H1, H2, H3);
        __syncthreads();
        layernorm_rows(H3, gs, bes);
        // segment_sum(e, j): atomic scatter-add (row-group local, no barrier needed)
        for (int row = tid >> 3; row < MT; row += RG) {
            if (vrow[row]) {
                const int jj = jrow[row];
                const float* r = H3 + (size_t)row * LAT;
                for (int c = sub; c < LAT; c += 8)
                    atomicAdd(&aggr[(size_t)jj * LAT + c], r[c]);
            }
        }
        __syncthreads();
    }
}

// ---------------- fused node MLP + residual (+ optional skip) ---------------
__global__ __launch_bounds__(TPB)
void node_mlp_kernel(const float* __restrict__ x, const float* __restrict__ aggr,
                     const float* __restrict__ w0, const float* __restrict__ b0,
                     const float* __restrict__ w1, const float* __restrict__ b1,
                     const float* __restrict__ w2, const float* __restrict__ b2,
                     const float* __restrict__ gamma, const float* __restrict__ beta,
                     float* __restrict__ out, const float* __restrict__ res2, int N) {
    const int Kp = K0P_NODE;             // 256
    extern __shared__ char smem[];
    bf16* W0f = (bf16*)smem;
    bf16* W1f = W0f + (size_t)Kp * LAT;
    bf16* W2f = W1f + (size_t)LAT * LAT;
    bf16* X   = W2f + (size_t)LAT * LAT;
    bf16* H1  = X + (size_t)MT * Kp;
    bf16* H2  = H1 + (size_t)MT * LAT;
    float* b0s = (float*)(H2 + (size_t)MT * LAT);
    float* b1s = b0s + LAT;
    float* b2s = b1s + LAT;
    float* gs  = b2s + LAT;
    float* bes = gs + LAT;
    float* H3  = bes + LAT;

    const int tid = threadIdx.x;
    stage_weight(W0f, w0, Kp, Kp);
    stage_weight(W1f, w1, LAT, LAT);
    stage_weight(W2f, w2, LAT, LAT);
    for (int c = tid; c < LAT; c += TPB) {
        b0s[c] = b0[c]; b1s[c] = b1[c]; b2s[c] = b2[c];
        gs[c] = gamma[c]; bes[c] = beta[c];
    }
    __syncthreads();

    const int ntiles = (N + MT - 1) / MT;
    const int sub = tid & 7;
    for (int t = blockIdx.x; t < ntiles; t += gridDim.x) {
        for (int row = tid >> 3; row < MT; row += RG) {
            const int n = t * MT + row;
            const bool valid = (n < N);
            bf16* xr = X + (size_t)row * Kp;
            for (int c = sub; c < Kp; c += 8) {
                float v = 0.f;
                if (valid) v = (c < LAT) ? x[(size_t)n * LAT + c]
                                         : aggr[(size_t)n * LAT + (c - LAT)];
                xr[c] = (bf16)v;
            }
        }
        __syncthreads();
        mlp3(X, Kp, W0f, W1f, W2f, b0s, b1s, b2s, H1, H2, H3);
        __syncthreads();
        layernorm_rows(H3, gs, bes);
        for (int row = tid >> 3; row < MT; row += RG) {
            const int n = t * MT + row;
            if (n < N) {
                const float* r = H3 + (size_t)row * LAT;
                for (int c = sub; c < LAT; c += 8) {
                    float v = r[c] + x[(size_t)n * LAT + c];          // + x residual
                    if (res2) v += res2[(size_t)n * LAT + c];          // + skip (up path)
                    out[(size_t)n * LAT + c] = v;
                }
            }
        }
        __syncthreads();
    }
}

// ------------------------------ plumbing kernels ----------------------------
__global__ void zero_kernel(float* p, long n) {
    long t = (long)blockIdx.x * blockDim.x + threadIdx.x;
    if (t < n) p[t] = 0.0f;
}
__global__ void deg_kernel(const int* __restrict__ i, float* deg, int E) {
    int e = blockIdx.x * blockDim.x + threadIdx.x;
    if (e < E) atomicAdd(&deg[i[e]], 1.0f);
}
__global__ void ws_kernel(const int* __restrict__ i, const int* __restrict__ j,
                          const float* __restrict__ w, const float* __restrict__ deg,
                          float* __restrict__ ws, float* __restrict__ aggr2, int E) {
    int e = blockIdx.x * blockDim.x + threadIdx.x;
    if (e < E) {
        int ii = i[e];
        float v = (w ? w[ii] : 1.0f) / deg[ii];
        ws[e] = v;
        atomicAdd(&aggr2[j[e]], v);
    }
}
__global__ void wfinal_kernel(const float* __restrict__ aggr2, float* wfull, int N) {
    int n = blockIdx.x * blockDim.x + threadIdx.x;
    if (n < N) wfull[n] = aggr2[n] + 1e-12f;
}
__global__ void ec_kernel(const float* __restrict__ ws, const int* __restrict__ j,
                          const float* __restrict__ wfull, float* ec, int E) {
    int e = blockIdx.x * blockDim.x + threadIdx.x;
    if (e < E) ec[e] = ws[e] / wfull[j[e]];
}
__global__ void wconv_kernel(const float* __restrict__ x, const int* __restrict__ src,
                             const int* __restrict__ tgt, const float* __restrict__ ew,
                             float* __restrict__ out, int E, int C) {
    long t = (long)blockIdx.x * blockDim.x + threadIdx.x;
    if (t < (long)E * C) {
        int e = (int)(t / C), c = (int)(t % C);
        atomicAdd(&out[(size_t)tgt[e] * C + c], ew[e] * x[(size_t)src[e] * C + c]);
    }
}
__global__ void gather_kernel(float* __restrict__ dst, const float* __restrict__ src,
                              const int* __restrict__ ids, int K, int C) {
    long t = (long)blockIdx.x * blockDim.x + threadIdx.x;
    if (t < (long)K * C) {
        int k = (int)(t / C), c = (int)(t % C);
        dst[t] = src[(size_t)ids[k] * C + c];
    }
}
__global__ void scatter_kernel(float* __restrict__ dst, const float* __restrict__ src,
                               const int* __restrict__ ids, int K, int C) {
    long t = (long)blockIdx.x * blockDim.x + threadIdx.x;
    if (t < (long)K * C) {
        int k = (int)(t / C), c = (int)(t % C);
        dst[(size_t)ids[k] * C + c] = src[t];
    }
}

// ------------------------------- host driver --------------------------------
struct Mlp { const float *b0, *b1, *b2, *be, *g, *w0, *w1, *w2; };
static Mlp get_mlp(void* const* d_in, int idx) {
    // jax tree order inside each MLP dict (alphabetical): b0,b1,b2,be,g,w0,w1,w2
    Mlp m;
    m.b0 = (const float*)d_in[idx + 0];
    m.b1 = (const float*)d_in[idx + 1];
    m.b2 = (const float*)d_in[idx + 2];
    m.be = (const float*)d_in[idx + 3];
    m.g  = (const float*)d_in[idx + 4];
    m.w0 = (const float*)d_in[idx + 5];
    m.w1 = (const float*)d_in[idx + 6];
    m.w2 = (const float*)d_in[idx + 7];
    return m;
}

static inline int grid1(long n, int tpb) { return (int)((n + tpb - 1) / tpb); }
static inline int clampi(int v, int hi) { return v < hi ? v : hi; }

static void zero(float* p, long n, hipStream_t s) {
    zero_kernel<<<grid1(n, 256), 256, 0, s>>>(p, n);
}

static void run_gmp(const Mlp& em, const Mlp& nm, const float* x, const float* pos,
                    const int* gi, const int* gj, int N, int E,
                    float* aggr, float* out, const float* res2, hipStream_t s) {
    zero(aggr, (long)N * LAT, s);
    int gE = clampi((E + MT - 1) / MT, 4096);
    hipFuncSetAttribute((const void*)edge_mlp_kernel,
                        hipFuncAttributeMaxDynamicSharedMemorySize, (int)smem_bytes(K0P_EDGE));
    edge_mlp_kernel<<<gE, TPB, smem_bytes(K0P_EDGE), s>>>(
        x, pos, gi, gj, em.w0, em.b0, em.w1, em.b1, em.w2, em.b2, em.g, em.be, aggr, E);
    int gN = clampi((N + MT - 1) / MT, 4096);
    hipFuncSetAttribute((const void*)node_mlp_kernel,
                        hipFuncAttributeMaxDynamicSharedMemorySize, (int)smem_bytes(K0P_NODE));
    node_mlp_kernel<<<gN, TPB, smem_bytes(K0P_NODE), s>>>(
        x, aggr, nm.w0, nm.b0, nm.w1, nm.b1, nm.w2, nm.b2, nm.g, nm.be, out, res2, N);
}

static void run_calew(const int* gi, const int* gj, const float* w_in, int N, int E,
                      float* deg, float* aggr2, float* wsbuf, float* wfull, float* ec,
                      hipStream_t s) {
    zero(deg, N, s);
    zero(aggr2, N, s);
    deg_kernel<<<grid1(E, 256), 256, 0, s>>>(gi, deg, E);
    ws_kernel<<<grid1(E, 256), 256, 0, s>>>(gi, gj, w_in, deg, wsbuf, aggr2, E);
    wfinal_kernel<<<grid1(N, 256), 256, 0, s>>>(aggr2, wfull, N);
    ec_kernel<<<grid1(E, 256), 256, 0, s>>>(wsbuf, gj, wfull, ec, E);
}

static void run_wconv(const float* x, const int* src, const int* tgt, const float* ew,
                      float* out, int N, int E, int C, hipStream_t s) {
    zero(out, (long)N * C, s);
    wconv_kernel<<<grid1((long)E * C, 256), 256, 0, s>>>(x, src, tgt, ew, out, E, C);
}

extern "C" void kernel_launch(void* const* d_in, const int* in_sizes, int n_in,
                              void* d_out, int out_size, void* d_ws, size_t ws_size,
                              hipStream_t stream) {
    (void)n_in; (void)out_size; (void)ws_size;
    // inputs: h, pos, g0, g1, g2, id0, id1, params(tree-flattened)
    const float* h_in  = (const float*)d_in[0];
    const float* pos_in = (const float*)d_in[1];
    const int N0 = in_sizes[0] / LAT;
    const int E0 = in_sizes[2] / 2;
    const int E1 = in_sizes[3] / 2;
    const int E2 = in_sizes[4] / 2;
    const int N1 = in_sizes[5];
    const int N2 = in_sizes[6];
    const int* g0 = (const int*)d_in[2]; const int *gi0 = g0, *gj0 = g0 + E0;
    const int* g1 = (const int*)d_in[3]; const int *gi1 = g1, *gj1 = g1 + E1;
    const int* g2 = (const int*)d_in[4]; const int *gi2 = g2, *gj2 = g2 + E2;
    const int* id0 = (const int*)d_in[5];
    const int* id1 = (const int*)d_in[6];
    // params tree order: bottom(edge,node), down[0..1](edge,node), up[0..1](edge,node)
    Mlp bot_e = get_mlp(d_in, 7),  bot_n = get_mlp(d_in, 15);
    Mlp d0_e  = get_mlp(d_in, 23), d0_n  = get_mlp(d_in, 31);
    Mlp d1_e  = get_mlp(d_in, 39), d1_n  = get_mlp(d_in, 47);
    Mlp u0_e  = get_mlp(d_in, 55), u0_n  = get_mlp(d_in, 63);
    Mlp u1_e  = get_mlp(d_in, 71), u1_n  = get_mlp(d_in, 79);

    // workspace carve (floats)
    float* p = (float*)d_ws;
    auto take = [&](long n) { float* r = p; p += n; return r; };
    float* douts0  = take((long)N0 * LAT);
    float* douts1  = take((long)N1 * LAT);
    float* aggr    = take((long)N0 * LAT);   // reused across all 5 GMPs
    float* hconv   = take((long)N0 * LAT);   // reused across all 4 wconv(h)
    float* h1buf   = take((long)N1 * LAT);
    float* h2buf   = take((long)N2 * LAT);
    float* hbot    = take((long)N2 * LAT);
    float* uph     = take((long)N0 * LAT);   // reused for both unpools
    float* hup1    = take((long)N1 * LAT);
    float* pos1buf = take((long)N1 * 3);     // dpos1
    float* pos2buf = take((long)N2 * 3);
    float* posconv = take((long)N0 * 3);     // reused
    float* w1buf   = take(N1);
    float* deg     = take(N0);
    float* aggr2   = take(N0);
    float* wfull   = take(N0);
    float* wsbuf   = take(E0);
    float* ec0     = take(E0);
    float* ec1     = take(E1);

    // ---- down level 0 ----
    run_gmp(d0_e, d0_n, h_in, pos_in, gi0, gj0, N0, E0, aggr, douts0, nullptr, stream);
    run_calew(gi0, gj0, nullptr, N0, E0, deg, aggr2, wsbuf, wfull, ec0, stream);
    run_wconv(douts0, gi0, gj0, ec0, hconv, N0, E0, LAT, stream);
    run_wconv(pos_in, gi0, gj0, ec0, posconv, N0, E0, 3, stream);
    gather_kernel<<<grid1((long)N1 * LAT, 256), 256, 0, stream>>>(h1buf, hconv, id0, N1, LAT);
    gather_kernel<<<grid1((long)N1 * 3, 256), 256, 0, stream>>>(pos1buf, posconv, id0, N1, 3);
    gather_kernel<<<grid1(N1, 256), 256, 0, stream>>>(w1buf, wfull, id0, N1, 1);

    // ---- down level 1 ----
    run_gmp(d1_e, d1_n, h1buf, pos1buf, gi1, gj1, N1, E1, aggr, douts1, nullptr, stream);
    run_calew(gi1, gj1, w1buf, N1, E1, deg, aggr2, wsbuf, wfull, ec1, stream);
    run_wconv(douts1, gi1, gj1, ec1, hconv, N1, E1, LAT, stream);
    run_wconv(pos1buf, gi1, gj1, ec1, posconv, N1, E1, 3, stream);
    gather_kernel<<<grid1((long)N2 * LAT, 256), 256, 0, stream>>>(h2buf, hconv, id1, N2, LAT);
    gather_kernel<<<grid1((long)N2 * 3, 256), 256, 0, stream>>>(pos2buf, posconv, id1, N2, 3);

    // ---- bottom ----
    run_gmp(bot_e, bot_n, h2buf, pos2buf, gi2, gj2, N2, E2, aggr, hbot, nullptr, stream);

    // ---- up level (d = 1) ----
    zero(uph, (long)N1 * LAT, stream);
    scatter_kernel<<<grid1((long)N2 * LAT, 256), 256, 0, stream>>>(uph, hbot, id1, N2, LAT);
    run_wconv(uph, gj1, gi1, ec1, hconv, N1, E1, LAT, stream);   // aggregating=False: src=j, tgt=i
    run_gmp(u0_e, u0_n, hconv, pos1buf, gi1, gj1, N1, E1, aggr, hup1, douts1, stream);

    // ---- up level (d = 0) ----
    zero(uph, (long)N0 * LAT, stream);
    scatter_kernel<<<grid1((long)N1 * LAT, 256), 256, 0, stream>>>(uph, hup1, id0, N1, LAT);
    run_wconv(uph, gj0, gi0, ec0, hconv, N0, E0, LAT, stream);
    run_gmp(u1_e, u1_n, hconv, pos_in, gi0, gj0, N0, E0, aggr, (float*)d_out, douts0, stream);
}
